// FISLayerParameterSharingV1_3719441679091
// MI455X (gfx1250) — matrix operand order repaired and checked by reading, not compiled, not run on hardware
//
#include <hip/hip_runtime.h>
#include <hip/hip_bf16.h>

// ---------------------------------------------------------------------------
// Problem constants (match reference setup_inputs()).
// ---------------------------------------------------------------------------
#define Bv  16
#define Cv  32
#define Hv  256
#define Wv  256
#define T2v 8
#define HWv (Hv * Wv)

typedef __attribute__((ext_vector_type(16))) __bf16 v16bf;
typedef __attribute__((ext_vector_type(8)))  float  v8f;

// ---------------------------------------------------------------------------
// Kernel 1: channel contraction via WMMA.
//   All six einsums "tc,bchw->bthw" are a single GEMM:
//     A = [48 trees x 32 channels]  (three 16x32 bf16 tiles)
//     B = [32 channels x 16 pixels] (one 32x16 bf16 tile, shared by 3 WMMAs)
//   K = 32 == C, so ONE v_wmma_f32_16x16x32_bf16 finishes a whole 16x16 tile.
// ---------------------------------------------------------------------------
__device__ inline v16bf load_alpha_pair(const float* __restrict__ aLo,
                                        const float* __restrict__ aHi,
                                        int lane) {
  // 16-bit A-matrix 16x32 layout (ISA 7.12.2):
  //  lanes 0-15  : M = lane,    V0..V3 -> K=0..7,   V4..V7 -> K=16..23
  //  lanes 16-31 : M = lane-16, V0..V3 -> K=8..15,  V4..V7 -> K=24..31
  const int hi = lane >> 4;
  const int m  = lane & 15;
  const float* a = (m < 8) ? aLo : aHi;   // rows 0-7 = first alpha, 8-15 = second
  const int t = m & 7;
  v16bf r;
#pragma unroll
  for (int e = 0; e < 16; ++e) {
    const int k = (e < 8) ? (e + (hi ? 8 : 0)) : ((e - 8) + (hi ? 24 : 16));
    r[e] = (__bf16)a[t * Cv + k];
  }
  return r;
}

__global__ void __launch_bounds__(256)
wmma_weight_kernel(const float* __restrict__ x,
                   const float* __restrict__ a1,  const float* __restrict__ a2,
                   const float* __restrict__ a3,  const float* __restrict__ a1p,
                   const float* __restrict__ a2p, const float* __restrict__ a3p,
                   float* __restrict__ fields /* 6 x [B,T2,H,W] */) {
  const int lane = threadIdx.x & 31;
  const int wave = threadIdx.x >> 5;
  const int tile = blockIdx.x * 8 + wave;      // one 16-pixel tile per wave
  const int tilesPerB = HWv / 16;
  const int b       = tile / tilesPerB;
  const int pixBase = (tile % tilesPerB) * 16;
  const int n       = lane & 15;               // pixel column N within tile
  const int hi      = lane >> 4;
  const int kOff    = hi ? 8 : 0;              // B-matrix K offset per lane group

  // --- B matrix: 32x16 bf16. lanes 0-15 hold K=0..7,16..23, lanes 16-31 K=8..15,24..31,
  //     two K values per VGPR. x is read exactly once in the whole pipeline,
  //     so use non-temporal loads (TH=NT) to keep it out of L2.
  v16bf bm;
  const float* xb = x + ((size_t)b * Cv) * HWv + pixBase + n;
#pragma unroll
  for (int i = 0; i < 16; ++i)
    bm[i] = (__bf16)__builtin_nontemporal_load(xb + (size_t)(kOff + i + (i < 8 ? 0 : 8)) * HWv);

  // --- A tiles: (a1,a2) (a3,a1p) (a2p,a3p) -> fields 0..5 in that pairing.
  const v16bf A0 = load_alpha_pair(a1,  a2,  lane);
  const v16bf A1 = load_alpha_pair(a3,  a1p, lane);
  const v16bf A2 = load_alpha_pair(a2p, a3p, lane);

  v8f c0 = {}, c1 = {}, c2 = {};
  c0 = __builtin_amdgcn_wmma_f32_16x16x32_bf16(false, A0, false, bm, (short)0, c0, false, false);
  c1 = __builtin_amdgcn_wmma_f32_16x16x32_bf16(false, A1, false, bm, (short)0, c1, false, false);
  c2 = __builtin_amdgcn_wmma_f32_16x16x32_bf16(false, A2, false, bm, (short)0, c2, false, false);

  // --- Scatter D. C/D layout: VGPR r -> M = r (lanes 0-15) or r+8 (lanes 16-31),
  //     N = lane%16. Row M maps to field = 2*g + hi, tree t = r.
  const size_t F = (size_t)Bv * T2v * HWv;     // elements per field
  const size_t pix = (size_t)pixBase + n;
#pragma unroll
  for (int r = 0; r < 8; ++r) {
    const size_t base = ((size_t)b * T2v + r) * HWv + pix;
    fields[(size_t)(0 + hi) * F + base] = c0[r];   // fields 0/1: W1, W2
    fields[(size_t)(2 + hi) * F + base] = c1[r];   // fields 2/3: W3, W1p
    fields[(size_t)(4 + hi) * F + base] = c2[r];   // fields 4/5: W2p, W3p
  }
}

// ---------------------------------------------------------------------------
// Kernel 2: strict directional 2D prefix sum of one 256x256 plane, fused with
// the pointwise multiply. The whole plane lives in LDS (256 x 257 x 4B =
// 257 KB — only possible because CDNA5 gives 320 KB LDS per WGP).
// Row stride 257 (odd) => conflict-free for both row-wise and column-wise
// serial sweeps across the 64 LDS banks.
// Staging uses GLOBAL_LOAD_ASYNC_TO_LDS_B128 (memory->LDS without a VGPR
// round trip, tracked on the dedicated ASYNCcnt), drained with
// s_wait_asynccnt before the workgroup barrier.
//   revH/revW select the strict (exclusive) scan direction per axis.
//   accumulate==0 : dst = mul * scan      accumulate==1 : dst *= scan
// ---------------------------------------------------------------------------
__global__ void __launch_bounds__(256)
plane_scan_kernel(const float* __restrict__ src,  // [B,T2,H,W]
                  const float* __restrict__ mul,  // [B,T2,H,W] (unused if accumulate)
                  float* __restrict__ dst,
                  int revH, int revW, int accumulate,
                  int dstTreeStride, int dstTreeOff) {
  extern __shared__ float lds[];                 // Hv * (Wv + 1) floats
  const int   plane = blockIdx.x;                // b*T2 + t
  const int   b = plane / T2v, t = plane % T2v;
  const size_t soff = (size_t)plane * HWv;
  const size_t doff = ((size_t)b * dstTreeStride + dstTreeOff + t) * HWv;
  const int tid = threadIdx.x;                   // 256 threads
  const int P = Wv + 1;                          // padded row stride

  // Stage plane into LDS: 16B async copies, 4-float chunks never cross the
  // padded row boundary (rows are 256 floats, chunks are w%4==0).
  for (int i4 = tid * 4; i4 < HWv; i4 += 256 * 4) {
    const int h = i4 >> 8, w = i4 & (Wv - 1);
    const unsigned ldsByte = (unsigned)((h * P + w) * sizeof(float));
    const float* g = src + soff + i4;
    asm volatile("global_load_async_to_lds_b128 %0, %1, off"
                 :: "v"(ldsByte), "v"(g)
                 : "memory");
  }
  asm volatile("s_wait_asynccnt 0" ::: "memory");
  __syncthreads();

  // Exclusive scan along W: one thread per row, serial over 256 elements.
  {
    float run = 0.f;
    float* row = lds + tid * P;
    if (revW) { for (int w = Wv - 1; w >= 0; --w) { const float v = row[w]; row[w] = run; run += v; } }
    else      { for (int w = 0; w < Wv;     ++w) { const float v = row[w]; row[w] = run; run += v; } }
  }
  __syncthreads();

  // Exclusive scan along H: one thread per column (stride-257 => no conflicts).
  {
    float run = 0.f;
    if (revH) { for (int h = Hv - 1; h >= 0; --h) { const float v = lds[h * P + tid]; lds[h * P + tid] = run; run += v; } }
    else      { for (int h = 0; h < Hv;     ++h) { const float v = lds[h * P + tid]; lds[h * P + tid] = run; run += v; } }
  }
  __syncthreads();

  // Fused pointwise emit. The mul field is single-use -> non-temporal load.
  for (int i = tid; i < HWv; i += 256) {
    const int h = i >> 8, w = i & (Wv - 1);
    const float s = lds[h * P + w];
    if (accumulate) dst[doff + i] *= s;
    else            dst[doff + i] = __builtin_nontemporal_load(mul + soff + i) * s;
  }
}

// ---------------------------------------------------------------------------
// Launch: 1 WMMA pass + 4 plane passes.
//   SW: h'>h, w'<w -> revH=1, revW=0
//   NE: h'<h, w'>w -> revH=0, revW=1
//   SE: h'>h, w'>w -> revH=1, revW=1
// ---------------------------------------------------------------------------
extern "C" void kernel_launch(void* const* d_in, const int* in_sizes, int n_in,
                              void* d_out, int out_size, void* d_ws, size_t ws_size,
                              hipStream_t stream) {
  const float* x   = (const float*)d_in[0];
  const float* a1  = (const float*)d_in[1];
  const float* a2  = (const float*)d_in[2];
  const float* a3  = (const float*)d_in[3];
  const float* a1p = (const float*)d_in[4];
  const float* a2p = (const float*)d_in[5];
  const float* a3p = (const float*)d_in[6];
  float* out    = (float*)d_out;          // [B, 16, H, W]
  float* fields = (float*)d_ws;           // 6 x [B, T2, H, W] f32
  const size_t F = (size_t)Bv * T2v * HWv;

  // Opt in to >64KB dynamic LDS (CDNA5 WGP has 320 KB).
  const int ldsBytes = Hv * (Wv + 1) * (int)sizeof(float);   // 263168 B
  (void)hipFuncSetAttribute((const void*)plane_scan_kernel,
                            hipFuncAttributeMaxDynamicSharedMemorySize, ldsBytes);

  // K1: all six weighted fields via WMMA. 65536 waves, 8 waves/block.
  const int tiles = Bv * HWv / 16;
  wmma_weight_kernel<<<tiles / 8, 256, 0, stream>>>(x, a1, a2, a3, a1p, a2p, a3p, fields);

  const int planes = Bv * T2v;  // 128 workgroups, one plane each

  // V2 = W2 * P_SW(W3)   (in place over the W3 field)
  plane_scan_kernel<<<planes, 256, ldsBytes, stream>>>(
      fields + 2 * F, fields + 1 * F, fields + 2 * F, 1, 0, 0, T2v, 0);

  // linear = W1 * P_NE(V2) -> out trees [0..7]
  plane_scan_kernel<<<planes, 256, ldsBytes, stream>>>(
      fields + 2 * F, fields + 0 * F, out, 0, 1, 0, 2 * T2v, 0);

  // cherry = W1p * P_SW(W3p) -> out trees [8..15]
  plane_scan_kernel<<<planes, 256, ldsBytes, stream>>>(
      fields + 5 * F, fields + 3 * F, out, 1, 0, 0, 2 * T2v, T2v);

  // cherry *= P_SE(W2p)
  plane_scan_kernel<<<planes, 256, ldsBytes, stream>>>(
      fields + 4 * F, nullptr, out, 1, 1, 1, 2 * T2v, T2v);
}
